// FuzzyDirGCN_77773267796195
// MI455X (gfx1250) — compile-verified, matrix operand order checked by reading, not run on hardware
//
#include <hip/hip_runtime.h>

typedef __attribute__((ext_vector_type(16))) _Float16 v16h;
typedef __attribute__((ext_vector_type(8)))  float    v8f;

#define DF   128      // feature dim (D_IN == HIDDEN == 128)
#define DK2  256      // fused K for layer GEMMs (agg_s | agg_d)
#define DOUT 64

static __host__ __device__ inline int cdiv(int a, int b) { return (a + b - 1) / b; }

// ---------------------------------------------------------------------------
// virtual edge list: i < E       : (s,r) = (snd[i], rcv[i]),   ow=cos^2, iw=sin^2
//                    E <= i < 2E : (s,r) = (rcv[e], snd[e]),   ow=sin^2, iw=cos^2
//                    i >= 2E     : self loop,                  ow=iw=1
// ---------------------------------------------------------------------------
__device__ __forceinline__ void edge_sr(int i, const int* __restrict__ snd,
                                        const int* __restrict__ rcv, int E,
                                        int& s, int& r) {
    if (i < E)            { s = snd[i];     r = rcv[i]; }
    else if (i < 2 * E)   { s = rcv[i - E]; r = snd[i - E]; }
    else                  { s = i - 2 * E;  r = s; }
}

__global__ void zero_kernel(float* __restrict__ p, long n) {
    long i = (long)blockIdx.x * blockDim.x + threadIdx.x;
    long stride = (long)gridDim.x * blockDim.x;
    for (; i < n; i += stride) p[i] = 0.0f;
}

// degree accumulation: deg_s[cs] += out_w ; deg_r[cs] += in_w   (both indexed by cs!)
__global__ void deg_kernel(const int* __restrict__ snd, const int* __restrict__ rcv,
                           const float* __restrict__ theta, int E, int Nn,
                           float* __restrict__ degS, float* __restrict__ degR) {
    int i = blockIdx.x * blockDim.x + threadIdx.x;
    int M = 2 * E + Nn;
    if (i >= M) return;
    int s, r; edge_sr(i, snd, rcv, E, s, r);
    float ow, iw;
    if (i < E)          { float c = cosf(theta[i]);     float sn = sinf(theta[i]);     ow = c * c;  iw = sn * sn; }
    else if (i < 2 * E) { float c = cosf(theta[i - E]); float sn = sinf(theta[i - E]); ow = sn * sn; iw = c * c; }
    else                { ow = 1.0f; iw = 1.0f; }
    unsafeAtomicAdd(degS + s, ow);
    unsafeAtomicAdd(degR + s, iw);
}

// in-place deg -> dis = (deg+1e-12 > 1e11) ? 0 : rsqrt(deg+1e-12)
__global__ void dis_kernel(float* __restrict__ d, int n) {
    int i = blockIdx.x * blockDim.x + threadIdx.x;
    if (i >= n) return;
    float v = d[i] + 1e-12f;
    d[i] = (v > 1e11f) ? 0.0f : rsqrtf(v);
}

// per-virtual-edge normalized weights
__global__ void weight_kernel(const int* __restrict__ snd, const int* __restrict__ rcv,
                              const float* __restrict__ theta, int E, int Nn,
                              const float* __restrict__ disS, const float* __restrict__ disR,
                              float* __restrict__ wS, float* __restrict__ wD) {
    int i = blockIdx.x * blockDim.x + threadIdx.x;
    int M = 2 * E + Nn;
    if (i >= M) return;
    int s, r; edge_sr(i, snd, rcv, E, s, r);
    float ow, iw;
    if (i < E)          { float c = cosf(theta[i]);     float sn = sinf(theta[i]);     ow = c * c;  iw = sn * sn; }
    else if (i < 2 * E) { float c = cosf(theta[i - E]); float sn = sinf(theta[i - E]); ow = sn * sn; iw = c * c; }
    else                { ow = 1.0f; iw = 1.0f; }
    wS[i] = disS[s] * ow * disR[r];
    wD[i] = disR[s] * iw * disS[r];
}

// one wave (32 lanes) per virtual edge; each lane handles 4 of 128 features.
// agg layout: [N][256] f32, cols 0..127 = agg_s, 128..255 = agg_d (fused GEMM input)
__global__ void scatter_kernel(const float* __restrict__ X,
                               const int* __restrict__ snd, const int* __restrict__ rcv,
                               const float* __restrict__ wS, const float* __restrict__ wD,
                               float* __restrict__ agg, int E, int Nn) {
    int wave = (blockIdx.x * blockDim.x + threadIdx.x) >> 5;
    int lane = threadIdx.x & 31;
    int M = 2 * E + Nn;
    if (wave >= M) return;
    int s, r; edge_sr(wave, snd, rcv, E, s, r);
    float a = wS[wave];
    float b = wD[wave];
    const float4 v = *(const float4*)(X + (long)s * DF + lane * 4);
    float* dst = agg + (long)r * DK2 + lane * 4;
    unsafeAtomicAdd(dst + 0,        v.x * a);
    unsafeAtomicAdd(dst + 1,        v.y * a);
    unsafeAtomicAdd(dst + 2,        v.z * a);
    unsafeAtomicAdd(dst + 3,        v.w * a);
    unsafeAtomicAdd(dst + DF + 0,   v.x * b);
    unsafeAtomicAdd(dst + DF + 1,   v.y * b);
    unsafeAtomicAdd(dst + DF + 2,   v.z * b);
    unsafeAtomicAdd(dst + DF + 3,   v.w * b);
}

// ---------------------------------------------------------------------------
// B pre-pack into fragment-major layout:
//   dst[((k0/32)*(NC/16) + tc)*32 + lane][j]  (16 halves contiguous per lane)
// where element (lane, j) of fragment = B[k][n],
//   k = k0 + (j<8?0:16) + 8*(lane>>4) + (j&7),  n = tc*16 + (lane&15)
// Layer variant fuses the [0.5*Ws ; 0.5*Wd] stack + f32->f16 convert.
// ---------------------------------------------------------------------------
__global__ void pack_cat(const float* __restrict__ Ws, const float* __restrict__ Wd,
                         _Float16* __restrict__ dst) {   // K=DK2, NC=DF
    int idx = blockIdx.x * blockDim.x + threadIdx.x;
    if (idx >= DK2 * DF) return;
    int j    = idx & 15;
    int lane = (idx >> 4) & 31;
    int t    = idx >> 9;                 // fragment id = k0t*(DF/16) + tc
    int k0t  = t / (DF / 16);
    int tc   = t % (DF / 16);
    int k = k0t * 32 + ((j < 8) ? 0 : 16) + (lane >> 4) * 8 + (j & 7);
    int n = tc * 16 + (lane & 15);
    float v = (k < DF) ? 0.5f * Ws[k * DF + n] : 0.5f * Wd[(k - DF) * DF + n];
    dst[idx] = (_Float16)v;
}

__global__ void pack_single(const float* __restrict__ W, _Float16* __restrict__ dst) {
    // K=DF, NC=DOUT
    int idx = blockIdx.x * blockDim.x + threadIdx.x;
    if (idx >= DF * DOUT) return;
    int j    = idx & 15;
    int lane = (idx >> 4) & 31;
    int t    = idx >> 9;
    int k0t  = t / (DOUT / 16);
    int tc   = t % (DOUT / 16);
    int k = k0t * 32 + ((j < 8) ? 0 : 16) + (lane >> 4) * 8 + (j & 7);
    int n = tc * 16 + (lane & 15);
    dst[idx] = (_Float16)W[k * DOUT + n];
}

__global__ void bias_combine(const float* __restrict__ bs, const float* __restrict__ bd,
                             float* __restrict__ out, int n) {
    int i = blockIdx.x * blockDim.x + threadIdx.x;
    if (i < n) out[i] = 0.5f * (bs[i] + bd[i]);
}

// ---------------------------------------------------------------------------
// WMMA GEMM: D[M x NC] = relu?( A[M x K](f32->f16) * Bp(packed f16) + bias )
// One wave per 16x16 tile. Mrows must be a multiple of 16 (50000 = 3125*16).
// A-fragment: two runs of 8 contiguous f32 per lane -> four float4 loads.
// B-fragment: one contiguous v16h (32B) load per lane from packed layout.
// C/D (f32, 8 VGPR): lane L -> n = L&15, vgpr i -> m = 8*(L>>4) + i.
// ---------------------------------------------------------------------------
template <int K, int NC, int RELU>
__global__ void wmma_gemm(const float* __restrict__ A, int Mrows,
                          const v16h* __restrict__ Bp, const float* __restrict__ bias,
                          float* __restrict__ D) {
    constexpr int colTiles = NC / 16;
    int wave = (blockIdx.x * blockDim.x + threadIdx.x) >> 5;
    int lane = threadIdx.x & 31;
    int rowTiles = Mrows >> 4;
    if (wave >= rowTiles * colTiles) return;
    int tr = wave / colTiles;
    int tc = wave % colTiles;
    int hi = lane >> 4;
    int arow = tr * 16 + (lane & 15);
    int ncol = tc * 16 + (lane & 15);

    const float4* Ar = (const float4*)(A + (long)arow * K);
    v8f c = {};
#pragma unroll
    for (int k0 = 0; k0 < K; k0 += 32) {
        int base = (k0 >> 2) + hi * 2;            // float4 index of k0 + 8*hi
        float4 p0 = Ar[base];
        float4 p1 = Ar[base + 1];
        float4 p2 = Ar[base + 4];                 // k0 + 16 + 8*hi
        float4 p3 = Ar[base + 5];
        v16h a;
        a[0]  = (_Float16)p0.x; a[1]  = (_Float16)p0.y;
        a[2]  = (_Float16)p0.z; a[3]  = (_Float16)p0.w;
        a[4]  = (_Float16)p1.x; a[5]  = (_Float16)p1.y;
        a[6]  = (_Float16)p1.z; a[7]  = (_Float16)p1.w;
        a[8]  = (_Float16)p2.x; a[9]  = (_Float16)p2.y;
        a[10] = (_Float16)p2.z; a[11] = (_Float16)p2.w;
        a[12] = (_Float16)p3.x; a[13] = (_Float16)p3.y;
        a[14] = (_Float16)p3.z; a[15] = (_Float16)p3.w;
        v16h b = Bp[((k0 >> 5) * colTiles + tc) * 32 + lane];
        c = __builtin_amdgcn_wmma_f32_16x16x32_f16(false, a, false, b,
                                                   (short)0, c, false, false);
    }
    float bv = bias[ncol];
    float* Drow = D + (long)(tr * 16 + hi * 8) * NC + ncol;
#pragma unroll
    for (int i = 0; i < 8; ++i) {
        float v = c[i] + bv;
        if (RELU) v = fmaxf(v, 0.0f);
        Drow[(long)i * NC] = v;                   // constant byte offsets
    }
}

// ---------------------------------------------------------------------------
extern "C" void kernel_launch(void* const* d_in, const int* in_sizes, int n_in,
                              void* d_out, int out_size, void* d_ws, size_t ws_size,
                              hipStream_t stream) {
    const float* x      = (const float*)d_in[0];
    const int*   edges  = (const int*)d_in[1];   // (2, E) flat, JAX default int32
    const float* theta  = (const float*)d_in[2];
    const float* W1s    = (const float*)d_in[3];
    const float* W1d    = (const float*)d_in[4];
    const float* b1s    = (const float*)d_in[5];
    const float* b1d    = (const float*)d_in[6];
    const float* W2s    = (const float*)d_in[7];
    const float* W2d    = (const float*)d_in[8];
    const float* b2s    = (const float*)d_in[9];
    const float* b2d    = (const float*)d_in[10];
    const float* Wro    = (const float*)d_in[11];
    const float* bro    = (const float*)d_in[12];

    const int N = in_sizes[0] / DF;     // 50000 (multiple of 16)
    const int E = in_sizes[2];          // 400000
    const int M = 2 * E + N;            // 850000 virtual edges
    const int* snd = edges;
    const int* rcv = edges + E;

    // workspace partition (256B aligned)
    char* ws = (char*)d_ws;
    size_t off = 0;
    auto alloc = [&](size_t bytes) { char* p = ws + off; off = (off + bytes + 255) & ~(size_t)255; return p; };
    float*    degS  = (float*)alloc((size_t)N * 4);
    float*    degR  = (float*)alloc((size_t)N * 4);
    float*    wS    = (float*)alloc((size_t)M * 4);
    float*    wD    = (float*)alloc((size_t)M * 4);
    float*    agg   = (float*)alloc((size_t)N * DK2 * 4);
    float*    x1    = (float*)alloc((size_t)N * DF * 4);
    float*    x2    = (float*)alloc((size_t)N * DF * 4);
    _Float16* Bp1   = (_Float16*)alloc((size_t)DK2 * DF * 2);
    _Float16* Bp2   = (_Float16*)alloc((size_t)DK2 * DF * 2);
    _Float16* Bpro  = (_Float16*)alloc((size_t)DF * DOUT * 2);
    float*    bias1 = (float*)alloc(DF * 4);
    float*    bias2 = (float*)alloc(DF * 4);

    const int T = 256;

    // --- preprocessing ---
    zero_kernel<<<512, T, 0, stream>>>(degS, N);
    zero_kernel<<<512, T, 0, stream>>>(degR, N);
    deg_kernel<<<cdiv(M, T), T, 0, stream>>>(snd, rcv, theta, E, N, degS, degR);
    dis_kernel<<<cdiv(N, T), T, 0, stream>>>(degS, N);
    dis_kernel<<<cdiv(N, T), T, 0, stream>>>(degR, N);
    weight_kernel<<<cdiv(M, T), T, 0, stream>>>(snd, rcv, theta, E, N, degS, degR, wS, wD);

    // --- weight/bias preparation (fragment-major f16 packing) ---
    pack_cat<<<cdiv(DK2 * DF, T), T, 0, stream>>>(W1s, W1d, Bp1);
    pack_cat<<<cdiv(DK2 * DF, T), T, 0, stream>>>(W2s, W2d, Bp2);
    pack_single<<<cdiv(DF * DOUT, T), T, 0, stream>>>(Wro, Bpro);
    bias_combine<<<1, DF, 0, stream>>>(b1s, b1d, bias1, DF);
    bias_combine<<<1, DF, 0, stream>>>(b2s, b2d, bias2, DF);

    const int scatterBlocks = cdiv(M * 32, T);
    const int layerTiles    = (N / 16) * (DF / 16);    // 3125 * 8
    const int roTiles       = (N / 16) * (DOUT / 16);  // 3125 * 4

    // --- layer 1 ---
    zero_kernel<<<4096, T, 0, stream>>>(agg, (long)N * DK2);
    scatter_kernel<<<scatterBlocks, T, 0, stream>>>(x, snd, rcv, wS, wD, agg, E, N);
    wmma_gemm<DK2, DF, 1><<<cdiv(layerTiles * 32, T), T, 0, stream>>>(
        agg, N, (const v16h*)Bp1, bias1, x1);

    // --- layer 2 ---
    zero_kernel<<<4096, T, 0, stream>>>(agg, (long)N * DK2);
    scatter_kernel<<<scatterBlocks, T, 0, stream>>>(x1, snd, rcv, wS, wD, agg, E, N);
    wmma_gemm<DK2, DF, 1><<<cdiv(layerTiles * 32, T), T, 0, stream>>>(
        agg, N, (const v16h*)Bp2, bias2, x2);

    // --- readout ---
    wmma_gemm<DF, DOUT, 0><<<cdiv(roTiles * 32, T), T, 0, stream>>>(
        x2, N, (const v16h*)Bpro, bro, (float*)d_out);
}
